// Informer_83219286327780
// MI455X (gfx1250) — matrix operand order repaired and checked by reading, not compiled, not run on hardware
//
#include <hip/hip_runtime.h>
#include <stdint.h>
#include <stddef.h>

// Informer forward for MI455X (gfx1250, wave32).
// Dense GEMMs: v_wmma_f32_16x16x32_f16 with async global->LDS double-buffered B tiles,
// chunk-level register pipelining (batch ds_loads -> back-to-back WMMAs).
// Workspace requirement: ~196 MB.

typedef __attribute__((ext_vector_type(16))) _Float16 v16h;
typedef __attribute__((ext_vector_type(8)))  _Float16 v8h;
typedef __attribute__((ext_vector_type(8)))  float    v8f;

#define NSAMP 35   // 5 * ceil(ln(512 or 1024)) = 35 for every attention call here

// ---------------- threefry2x32 (exact JAX implementation) ----------------
__host__ __device__ static inline unsigned rotl32(unsigned x, int n) {
  return (x << n) | (x >> (32 - n));
}
__host__ __device__ static inline void threefry2x32(unsigned k0, unsigned k1,
                                                    unsigned x0, unsigned x1,
                                                    unsigned* r0, unsigned* r1) {
  const unsigned ks2 = 0x1BD11BDAu ^ k0 ^ k1;
  x0 += k0; x1 += k1;
#define TFR(r) { x0 += x1; x1 = rotl32(x1,(r)); x1 ^= x0; }
  TFR(13) TFR(15) TFR(26) TFR(6)   x0 += k1;  x1 += ks2 + 1u;
  TFR(17) TFR(29) TFR(16) TFR(24)  x0 += ks2; x1 += k0  + 2u;
  TFR(13) TFR(15) TFR(26) TFR(6)   x0 += k0;  x1 += k1  + 3u;
  TFR(17) TFR(29) TFR(16) TFR(24)  x0 += k1;  x1 += ks2 + 4u;
  TFR(13) TFR(15) TFR(26) TFR(6)   x0 += ks2; x1 += k0  + 5u;
#undef TFR
  *r0 = x0; *r1 = x1;
}

// ---------------- WMMA GEMM: C[M,N] = A[M,K] * W[N,K]^T + bias ----------------
// 256 threads = 8 waves; block tile 128x64; each wave computes 16 rows x 64 cols.
// B tile (shared by all 8 waves) is staged into LDS with async global->LDS DMA,
// double-buffered so the DMA overlaps the WMMA stream. A is loaded directly.
// Per chunk: gather ALL fragments into registers first, then 8 WMMAs back-to-back.
#define KC 64        // K-chunk per stage (2 WMMA K-steps)
#define BSTRIDE 72   // LDS row stride in halfs (144B = 9*16B: aligned, bank-spread)

__global__ __launch_bounds__(256)
void k_gemm_wmma(const _Float16* __restrict__ A, const _Float16* __restrict__ W,
                 const float* __restrict__ bias, float* __restrict__ C,
                 int M, int N, int K) {
  __shared__ _Float16 Bs[2][64 * BSTRIDE];   // 2 x 9216B

  const int tid  = threadIdx.x;
  const int wave = tid >> 5;
  const int lane = tid & 31;
  const int r    = lane & 15;
  const int hs   = lane >> 4;                 // half-wave select
  const int m0   = blockIdx.y * 128 + wave * 16;
  const int n0   = blockIdx.x * 64;

  // A fragment base: lane holds two 8-element K chunks at hs*8 and 16+hs*8
  const _Float16* Ap = A + (size_t)(m0 + r) * K + hs * 8;

  // async-stage one 64(n) x KC(k) B chunk into LDS buffer `buf`
  auto stage = [&](int buf, int kc) {
    // 64 rows * KC halfs = 512 x 16B segments; 256 threads x 2
    for (int s = tid; s < 512; s += 256) {
      int row = s >> 3, seg = s & 7;
      const _Float16* gsrc = W + (size_t)(n0 + row) * K + kc + seg * 8;
      uint32_t la = (uint32_t)(uintptr_t)&Bs[buf][row * BSTRIDE + seg * 8];
      asm volatile("global_load_async_to_lds_b128 %0, %1, off"
                   :: "v"(la), "v"((uint64_t)(uintptr_t)gsrc) : "memory");
    }
  };

  v8f acc[4] = {};
  stage(0, 0);
  int cur = 0;
  for (int kc = 0; kc < K; kc += KC) {
    asm volatile("s_wait_asynccnt 0x0" ::: "memory");
    __syncthreads();                      // all waves' DMA visible in LDS
    if (kc + KC < K) stage(cur ^ 1, kc + KC);   // overlap next DMA with compute

    // ---- gather all fragments for this chunk into registers ----
    v16h afr[2];
    v16h bfr[2][4];
#pragma unroll
    for (int kk2 = 0; kk2 < 2; ++kk2) {
      const int kk = kk2 * 32;
      v8h alo = *(const v8h*)(Ap + kc + kk);
      v8h ahi = *(const v8h*)(Ap + kc + kk + 16);
      afr[kk2] = __builtin_shufflevector(alo, ahi, 0,1,2,3,4,5,6,7,8,9,10,11,12,13,14,15);
#pragma unroll
      for (int j = 0; j < 4; ++j) {
        const _Float16* bp = &Bs[cur][(16 * j + r) * BSTRIDE + kk + hs * 8];
        v8h bl = *(const v8h*)bp;
        v8h bh = *(const v8h*)(bp + 16);
        bfr[kk2][j] = __builtin_shufflevector(bl, bh, 0,1,2,3,4,5,6,7,8,9,10,11,12,13,14,15);
      }
    }
    // ---- 8 back-to-back WMMAs (no D->A/B hazards: distinct accumulators) ----
#pragma unroll
    for (int kk2 = 0; kk2 < 2; ++kk2) {
#pragma unroll
      for (int j = 0; j < 4; ++j) {
        acc[j] = __builtin_amdgcn_wmma_f32_16x16x32_f16(false, afr[kk2], false, bfr[kk2][j],
                                                        (short)0, acc[j], false, false);
      }
    }
    __syncthreads();                      // reads done before buffer is restaged
    cur ^= 1;
  }

  const float bb0 = bias[n0 +  0 + r];
  const float bb1 = bias[n0 + 16 + r];
  const float bb2 = bias[n0 + 32 + r];
  const float bb3 = bias[n0 + 48 + r];
  // C/D layout: VGPR i of lane (hs,r) -> row m0 + i + 8*hs, col n0 + 16*j + r
#pragma unroll
  for (int i = 0; i < 8; ++i) {
    float* Cr = C + (size_t)(m0 + i + 8 * hs) * N + n0 + r;
    Cr[ 0] = acc[0][i] + bb0;
    Cr[16] = acc[1][i] + bb1;
    Cr[32] = acc[2][i] + bb2;
    Cr[48] = acc[3][i] + bb3;
  }
}

// ---------------- elementwise / small kernels ----------------
__global__ void k_f32_to_f16(const float* __restrict__ s, _Float16* __restrict__ d, int n) {
  int i = blockIdx.x * 256 + threadIdx.x;
  if (i < n) d[i] = (_Float16)s[i];
}

__global__ void k_gelu(float* __restrict__ x, int n) {
  int i = blockIdx.x * 256 + threadIdx.x;
  if (i < n) { float v = x[i]; x[i] = 0.5f * v * (1.f + erff(v * 0.70710678118654752f)); }
}

// input projection + positional embedding. dec_mode: rows l<512 take x_enc[b,512+l], else zeros.
__global__ __launch_bounds__(512)
void k_proj_emb(const float* __restrict__ xe, const float* __restrict__ W,
                const float* __restrict__ bb, const float* __restrict__ emb,
                float* __restrict__ X, int dec_mode) {
  int row = blockIdx.x;                 // 0..8191  (b*1024+l)
  int l = row & 1023, b = row >> 10;
  int n = threadIdx.x;
  float acc = bb[n] + emb[l * 512 + n];
  const float* src = nullptr;
  if (!dec_mode) src = xe + (size_t)row * 7;
  else if (l < 512) src = xe + (size_t)(b * 1024 + 512 + l) * 7;
  if (src) {
#pragma unroll
    for (int k = 0; k < 7; k++) acc += src[k] * W[n * 7 + k];
  }
  X[(size_t)row * 512 + n] = acc;
}

// layernorm over 512, optional residual add; out = LN(x + res)*g + b (in-place safe)
__global__ __launch_bounds__(256)
void k_ln(const float* __restrict__ x, const float* __restrict__ res,
          const float* __restrict__ g, const float* __restrict__ bta,
          float* __restrict__ out, int rows) {
  __shared__ float s1[256], s2[256];
  int row = blockIdx.x; int t = threadIdx.x;
  const float* xr = x + (size_t)row * 512;
  float v0 = xr[t]       + (res ? res[(size_t)row * 512 + t]       : 0.f);
  float v1 = xr[t + 256] + (res ? res[(size_t)row * 512 + t + 256] : 0.f);
  s1[t] = v0 + v1; s2[t] = v0 * v0 + v1 * v1;
  __syncthreads();
  for (int o = 128; o > 0; o >>= 1) { if (t < o) { s1[t] += s1[t + o]; s2[t] += s2[t + o]; } __syncthreads(); }
  float mu = s1[0] / 512.f;
  float var = s2[0] / 512.f - mu * mu;
  float rstd = rsqrtf(var + 1e-5f);
  out[(size_t)row * 512 + t]       = (v0 - mu) * rstd * g[t] + bta[t];
  out[(size_t)row * 512 + t + 256] = (v1 - mu) * rstd * g[t + 256] + bta[t + 256];
}

// circular-pad gather: X3[row, t*512+ch] = x[b, (l-1+t) mod 1024, ch]  (f16)
__global__ void k_build_x3(const float* __restrict__ x, _Float16* __restrict__ d) {
  int i = blockIdx.x * 256 + threadIdx.x;
  if (i >= 8192 * 1536) return;
  int row = i / 1536, c = i % 1536, t = c / 512, ch = c % 512;
  int b = row >> 10, l = row & 1023;
  int ls = (l - 1 + t + 1024) & 1023;
  d[i] = (_Float16)x[(size_t)(b * 1024 + ls) * 512 + ch];
}

// Wc (O=512, I=512, T=3) -> f16 [O, T*512 + I]
__global__ void k_repack_convw(const float* __restrict__ Wc, _Float16* __restrict__ d) {
  int i = blockIdx.x * 256 + threadIdx.x;
  if (i >= 512 * 1536) return;
  int o = i / 1536, c = i % 1536, t = c / 512, ch = c % 512;
  d[i] = (_Float16)Wc[(o * 512 + ch) * 3 + t];
}

// per-channel batchnorm stats over 8192 (B*L) rows
__global__ __launch_bounds__(256)
void k_bn_stats(const float* __restrict__ y, float* __restrict__ stats) {
  __shared__ float s1[256], s2[256];
  int c = blockIdx.x; int t = threadIdx.x;
  float a = 0.f, b2 = 0.f;
  for (int r = t; r < 8192; r += 256) { float v = y[(size_t)r * 512 + c]; a += v; b2 += v * v; }
  s1[t] = a; s2[t] = b2; __syncthreads();
  for (int o = 128; o > 0; o >>= 1) { if (t < o) { s1[t] += s1[t + o]; s2[t] += s2[t + o]; } __syncthreads(); }
  if (t == 0) { float mu = s1[0] / 8192.f; stats[c] = mu; stats[512 + c] = s2[0] / 8192.f - mu * mu; }
}

// BN -> ELU -> maxpool(k=3, s=2, -inf pad). out rows = (b, lo in 0..511)
__global__ __launch_bounds__(512)
void k_bn_elu_pool(const float* __restrict__ y, const float* __restrict__ stats,
                   const float* __restrict__ g, const float* __restrict__ bb,
                   float* __restrict__ out) {
  int row = blockIdx.x;                 // 0..4095
  int c = threadIdx.x;
  int b = row >> 9, lo = row & 511;
  float mu = stats[c], rstd = rsqrtf(stats[512 + c] + 1e-5f);
  float m = -3.4e38f;
  for (int dl = 0; dl < 3; dl++) {
    int l = 2 * lo - 1 + dl;
    if (l >= 0 && l < 1024) {
      float v = y[(size_t)(b * 1024 + l) * 512 + c];
      v = (v - mu) * rstd * g[c] + bb[c];
      v = (v > 0.f) ? v : expm1f(v);
      m = fmaxf(m, v);
    }
  }
  out[(size_t)row * 512 + c] = m;
}

// idx = threefry counter-mode bits & (Lk-1); exact JAX ordering (x0=i, x1=half+i)
__global__ void k_make_idx(unsigned k0, unsigned k1, int half, int mask, int* __restrict__ idx) {
  int i = blockIdx.x * 256 + threadIdx.x;
  if (i >= half) return;
  unsigned o0, o1;
  threefry2x32(k0, k1, (unsigned)i, (unsigned)(half + i), &o0, &o1);
  idx[i]        = (int)(o0 & (unsigned)mask);
  idx[half + i] = (int)(o1 & (unsigned)mask);
}

// M[b,h,q] = max_s(q.k_sample) - sum_s(q.k_sample)/Lk
__global__ __launch_bounds__(256)
void k_sample_M(const float* __restrict__ Q, const float* __restrict__ Km,
                const int* __restrict__ idx, float* __restrict__ Mout, int Lq, int Lk) {
  int tid = blockIdx.x * 256 + threadIdx.x;
  if (tid >= 64 * Lq) return;
  int q = tid % Lq; int bh = tid / Lq; int h = bh & 7; int b = bh >> 3;
  const float* qp = Q + (size_t)(b * Lq + q) * 512 + h * 64;
  float qv[64];
#pragma unroll
  for (int j = 0; j < 64; j++) qv[j] = qp[j];
  float mx = -3.4e38f, sm = 0.f;
  for (int s = 0; s < NSAMP; s++) {
    int ki = idx[q * NSAMP + s];
    const float* kp = Km + (size_t)(b * Lk + ki) * 512 + h * 64;
    float d = 0.f;
#pragma unroll
    for (int j = 0; j < 64; j++) d += qv[j] * kp[j];
    mx = fmaxf(mx, d); sm += d;
  }
  Mout[(size_t)bh * Lq + q] = mx - sm / (float)Lk;
}

// stable top-35 (ties -> lower index), one block per (b,h)
__global__ void k_topk(const float* __restrict__ Mbuf, int* __restrict__ mtop, int Lq) {
  __shared__ float sm[1024];
  int bh = blockIdx.x;
  for (int i = threadIdx.x; i < Lq; i += blockDim.x) sm[i] = Mbuf[(size_t)bh * Lq + i];
  __syncthreads();
  if (threadIdx.x == 0) {
    for (int u = 0; u < NSAMP; u++) {
      float best = -3.4e38f; int bi = 0;
      for (int i = 0; i < Lq; i++) if (sm[i] > best) { best = sm[i]; bi = i; }
      mtop[bh * NSAMP + u] = bi;
      sm[bi] = -3.4e38f;
    }
  }
}

// per selected query: full scores over Lk, causal mask, softmax, attn@V
__global__ __launch_bounds__(256)
void k_scores_av(const float* __restrict__ Q, const float* __restrict__ Km,
                 const float* __restrict__ V, const int* __restrict__ mtop,
                 float* __restrict__ AV, int Lq, int Lk, int masked) {
  __shared__ float sc[1024];
  __shared__ float qs[64];
  __shared__ float red[256];
  int u = blockIdx.x % NSAMP; int h = (blockIdx.x / NSAMP) & 7; int b = blockIdx.x / (NSAMP * 8);
  int t = threadIdx.x;
  int qi = mtop[(b * 8 + h) * NSAMP + u];
  if (t < 64) qs[t] = Q[(size_t)(b * Lq + qi) * 512 + h * 64 + t];
  __syncthreads();
  float lmax = -3.4e38f;
  for (int k = t; k < Lk; k += 256) {
    const float* kr = Km + (size_t)(b * Lk + k) * 512 + h * 64;
    float d = 0.f;
#pragma unroll
    for (int j = 0; j < 64; j++) d += qs[j] * kr[j];
    d *= 0.125f;                       // 1/sqrt(64)
    if (masked && k > qi) d = -3.0e38f;
    sc[k] = d; lmax = fmaxf(lmax, d);
  }
  red[t] = lmax; __syncthreads();
  for (int o = 128; o > 0; o >>= 1) { if (t < o) red[t] = fmaxf(red[t], red[t + o]); __syncthreads(); }
  float mx = red[0]; __syncthreads();
  float lsum = 0.f;
  for (int k = t; k < Lk; k += 256) { float e = expf(sc[k] - mx); sc[k] = e; lsum += e; }
  red[t] = lsum; __syncthreads();
  for (int o = 128; o > 0; o >>= 1) { if (t < o) red[t] += red[t + o]; __syncthreads(); }
  float inv = 1.f / red[0];
  if (t < 64) {
    float acc = 0.f;
    for (int k = 0; k < Lk; k++) acc += sc[k] * V[(size_t)(b * Lk + k) * 512 + h * 64 + t];
    AV[(size_t)((b * 8 + h) * NSAMP + u) * 64 + t] = acc * inv;
  }
}

// non-masked context: broadcast mean of V rows
__global__ __launch_bounds__(256)
void k_ctx_meanfill(const float* __restrict__ V, float* __restrict__ CTX, int Lq, int Lk) {
  __shared__ float mean[64];
  int h = blockIdx.x & 7, b = blockIdx.x >> 3; int t = threadIdx.x;
  if (t < 64) {
    float s = 0.f;
    for (int k = 0; k < Lk; k++) s += V[(size_t)(b * Lk + k) * 512 + h * 64 + t];
    mean[t] = s / (float)Lk;
  }
  __syncthreads();
  for (int i = t; i < Lq * 64; i += 256) {
    int q = i >> 6, d = i & 63;
    CTX[(size_t)(b * Lq + q) * 512 + h * 64 + d] = mean[d];
  }
}

// masked context: inclusive cumsum of V along sequence
__global__ void k_ctx_cumsum(const float* __restrict__ V, float* __restrict__ CTX, int Lq) {
  int b = blockIdx.x >> 3, h = blockIdx.x & 7, d = threadIdx.x;
  float run = 0.f;
  for (int q = 0; q < Lq; q++) {
    size_t o = (size_t)(b * Lq + q) * 512 + h * 64 + d;
    run += V[o];
    CTX[o] = run;
  }
}

__global__ void k_scatter_av(const float* __restrict__ AV, const int* __restrict__ mtop,
                             float* __restrict__ CTX, int Lq) {
  int tid = blockIdx.x * 256 + threadIdx.x;
  if (tid >= 64 * NSAMP * 64) return;
  int d = tid & 63; int u = (tid >> 6) % NSAMP; int bh = tid / (64 * NSAMP);
  int h = bh & 7, b = bh >> 3;
  int qi = mtop[bh * NSAMP + u];
  CTX[(size_t)(b * Lq + qi) * 512 + h * 64 + d] = AV[(size_t)(bh * NSAMP + u) * 64 + d];
}

// out[b,t] = dot(d[b,512+t,:], W_out[0,:]) + b_out[0]
__global__ __launch_bounds__(256)
void k_final_out(const float* __restrict__ dd, const float* __restrict__ Wo,
                 const float* __restrict__ bo, float* __restrict__ out) {
  __shared__ float red[256];
  int b = blockIdx.x >> 9, t = blockIdx.x & 511;
  size_t row = (size_t)(b * 1024 + 512 + t);
  int tt = threadIdx.x;
  float s = dd[row * 512 + tt] * Wo[tt] + dd[row * 512 + tt + 256] * Wo[tt + 256];
  red[tt] = s; __syncthreads();
  for (int o = 128; o > 0; o >>= 1) { if (tt < o) red[tt] += red[tt + o]; __syncthreads(); }
  if (tt == 0) out[b * 512 + t] = red[0] + bo[0];
}

// ==================================================================
extern "C" void kernel_launch(void* const* d_in, const int* in_sizes, int n_in,
                              void* d_out, int out_size, void* d_ws, size_t ws_size,
                              hipStream_t stream) {
  (void)in_sizes; (void)n_in; (void)out_size; (void)ws_size;
  int p = 0;
  auto nxt = [&]() { return (const float*)d_in[p++]; };

  struct AttnP { const float *qW,*qb,*kW,*kb,*vW,*vb,*oW,*ob; };
  auto getAttn = [&]() { AttnP a; a.qW=nxt(); a.qb=nxt(); a.kW=nxt(); a.kb=nxt();
                         a.vW=nxt(); a.vb=nxt(); a.oW=nxt(); a.ob=nxt(); return a; };
  struct EncL { AttnP a; const float *f1W,*f1b,*f2W,*f2b,*l1g,*l1b,*l2g,*l2b; };
  auto getEnc = [&]() { EncL e; e.a=getAttn(); e.f1W=nxt(); e.f1b=nxt(); e.f2W=nxt(); e.f2b=nxt();
                        e.l1g=nxt(); e.l1b=nxt(); e.l2g=nxt(); e.l2b=nxt(); return e; };

  // leaves in setup_inputs() insertion order (75 total)
  const float* enc_proj_W = nxt(); const float* enc_proj_b = nxt();
  const float* dec_proj_W = nxt(); const float* dec_proj_b = nxt();
  const float* enc_emb = nxt();    const float* dec_emb = nxt();
  EncL enc0 = getEnc(); EncL enc1 = getEnc();
  const float* conv_Wc = nxt(); const float* conv_bc = nxt();
  const float* bn_g = nxt();    const float* bn_b = nxt();
  const float* enc_ng = nxt();  const float* enc_nb = nxt();
  AttnP dself = getAttn(); AttnP dcross = getAttn();
  const float* d_f1W = nxt(); const float* d_f1b = nxt();
  const float* d_f2W = nxt(); const float* d_f2b = nxt();
  const float* d_l1g = nxt(); const float* d_l1b = nxt();
  const float* d_l2g = nxt(); const float* d_l2b = nxt();
  const float* d_l3g = nxt(); const float* d_l3b = nxt();
  const float* dec_ng = nxt(); const float* dec_nb = nxt();
  const float* out_W = nxt();  const float* out_b = nxt();
  const float* x_enc = nxt();

  // ---------------- workspace arena (~196 MB) ----------------
  char* w = (char*)d_ws;
  size_t off = 0;
  auto alloc = [&](size_t bytes) { size_t o = off; off += (bytes + 255) & ~(size_t)255; return o; };
  float* X   = (float*)(w + alloc((size_t)8192 * 512 * 4));
  float* Qb  = (float*)(w + alloc((size_t)8192 * 512 * 4));
  float* Kb  = (float*)(w + alloc((size_t)8192 * 512 * 4));
  float* Vb  = (float*)(w + alloc((size_t)8192 * 512 * 4));
  float* T0  = (float*)(w + alloc((size_t)8192 * 512 * 4));
  float* FF  = (float*)(w + alloc((size_t)8192 * 2048 * 4));
  float* ENC = (float*)(w + alloc((size_t)4096 * 512 * 4));
  _Float16* HA = (_Float16*)(w + alloc((size_t)8192 * 2048 * 2));
  _Float16* HW = (_Float16*)(w + alloc((size_t)2048 * 512 * 2));
  int*   IDX  = (int*)(w + alloc((size_t)1024 * NSAMP * 4));
  float* MBUF = (float*)(w + alloc((size_t)64 * 1024 * 4));
  int*   MTOP = (int*)(w + alloc((size_t)64 * NSAMP * 4));
  float* AV   = (float*)(w + alloc((size_t)64 * NSAMP * 64 * 4));
  float* BNS  = (float*)(w + alloc((size_t)1024 * 4));

  auto toh = [&](const float* s, _Float16* d, int n) {
    k_f32_to_f16<<<dim3((unsigned)((n + 255) / 256)), 256, 0, stream>>>(s, d, n);
  };
  auto gemm = [&](const _Float16* Af, const float* Wf, const float* bias, float* Cc,
                  int M, int N, int K) {
    toh(Wf, HW, N * K);
    k_gemm_wmma<<<dim3((unsigned)(N / 64), (unsigned)(M / 128)), 256, 0, stream>>>(Af, HW, bias, Cc, M, N, K);
  };
  auto attention = [&](const float* Qp, const float* Kp, const float* Vp, float* CTX,
                       int Lq, int Lk, int masked, int ctr) {
    unsigned fk0, fk1;
    threefry2x32(0u, 1234u, 0u, (unsigned)ctr, &fk0, &fk1);   // fold_in(key(1234), ctr)
    int n = Lq * NSAMP, half = n / 2;
    k_make_idx<<<dim3((unsigned)((half + 255) / 256)), 256, 0, stream>>>(fk0, fk1, half, Lk - 1, IDX);
    k_sample_M<<<dim3((unsigned)((64 * Lq) / 256)), 256, 0, stream>>>(Qp, Kp, IDX, MBUF, Lq, Lk);
    k_topk<<<dim3(64), 256, 0, stream>>>(MBUF, MTOP, Lq);
    k_scores_av<<<dim3(64 * NSAMP), 256, 0, stream>>>(Qp, Kp, Vp, MTOP, AV, Lq, Lk, masked);
    if (masked) k_ctx_cumsum<<<dim3(64), 64, 0, stream>>>(Vp, CTX, Lq);
    else        k_ctx_meanfill<<<dim3(64), 256, 0, stream>>>(Vp, CTX, Lq, Lk);
    k_scatter_av<<<dim3((unsigned)((64 * NSAMP * 64 + 255) / 256)), 256, 0, stream>>>(AV, MTOP, CTX, Lq);
  };
  auto attn_block = [&](const AttnP& A2, float* xio, const float* xkv,
                        int Mq, int Mkv, int Lq, int Lk, int masked, int ctr,
                        const float* lng, const float* lnb) {
    toh(xio, HA, Mq * 512);
    gemm(HA, A2.qW, A2.qb, Qb, Mq, 512, 512);
    if (xkv != (const float*)xio) toh(xkv, HA, Mkv * 512);
    gemm(HA, A2.kW, A2.kb, Kb, Mkv, 512, 512);
    gemm(HA, A2.vW, A2.vb, Vb, Mkv, 512, 512);
    attention(Qb, Kb, Vb, T0, Lq, Lk, masked, ctr);
    toh(T0, HA, Mq * 512);
    gemm(HA, A2.oW, A2.ob, Qb, Mq, 512, 512);
    k_ln<<<dim3((unsigned)Mq), 256, 0, stream>>>(xio, Qb, lng, lnb, xio, Mq);
  };
  auto ff_block = [&](const float* f1W, const float* f1b, const float* f2W, const float* f2b,
                      const float* lng, const float* lnb, float* Xc, int M) {
    toh(Xc, HA, M * 512);
    gemm(HA, f1W, f1b, FF, M, 2048, 512);
    k_gelu<<<dim3((unsigned)((M * 2048) / 256)), 256, 0, stream>>>(FF, M * 2048);
    toh(FF, HA, M * 2048);
    gemm(HA, f2W, f2b, T0, M, 512, 2048);
    k_ln<<<dim3((unsigned)M), 256, 0, stream>>>(Xc, T0, lng, lnb, Xc, M);
  };

  // ---------------- encoder ----------------
  k_proj_emb<<<dim3(8192), 512, 0, stream>>>(x_enc, enc_proj_W, enc_proj_b, enc_emb, X, 0);
  attn_block(enc0.a, X, X, 8192, 8192, 1024, 1024, 0, /*ctr*/1, enc0.l1g, enc0.l1b);
  ff_block(enc0.f1W, enc0.f1b, enc0.f2W, enc0.f2b, enc0.l2g, enc0.l2b, X, 8192);

  // conv distilling: circular conv(k=3) as GEMM K=1536, then BN -> ELU -> maxpool s=2
  k_build_x3<<<dim3((8192 * 1536) / 256), 256, 0, stream>>>(X, HA);
  k_repack_convw<<<dim3((512 * 1536) / 256), 256, 0, stream>>>(conv_Wc, HW);
  k_gemm_wmma<<<dim3(512 / 64, 8192 / 128), 256, 0, stream>>>(HA, HW, conv_bc, T0, 8192, 512, 1536);
  k_bn_stats<<<dim3(512), 256, 0, stream>>>(T0, BNS);
  k_bn_elu_pool<<<dim3(4096), 512, 0, stream>>>(T0, BNS, bn_g, bn_b, X);

  attn_block(enc1.a, X, X, 4096, 4096, 512, 512, 0, /*ctr*/2, enc1.l1g, enc1.l1b);
  ff_block(enc1.f1W, enc1.f1b, enc1.f2W, enc1.f2b, enc1.l2g, enc1.l2b, X, 4096);
  k_ln<<<dim3(4096), 256, 0, stream>>>(X, nullptr, enc_ng, enc_nb, ENC, 4096);

  // ---------------- decoder ----------------
  k_proj_emb<<<dim3(8192), 512, 0, stream>>>(x_enc, dec_proj_W, dec_proj_b, dec_emb, X, 1);
  attn_block(dself,  X, X,   8192, 8192, 1024, 1024, 1, /*ctr*/3, d_l1g, d_l1b);
  attn_block(dcross, X, ENC, 8192, 4096, 1024, 512,  0, /*ctr*/4, d_l2g, d_l2b);
  ff_block(d_f1W, d_f1b, d_f2W, d_f2b, d_l3g, d_l3b, X, 8192);
  k_ln<<<dim3(8192), 256, 0, stream>>>(X, nullptr, dec_ng, dec_nb, T0, 8192);
  k_final_out<<<dim3(4096), 256, 0, stream>>>(T0, out_W, out_b, (float*)d_out);
}
